// EPDeepseekMoE_30056181137960
// MI455X (gfx1250) — matrix operand order repaired and checked by reading, not compiled, not run on hardware
//
#include <hip/hip_runtime.h>

// Problem constants (from the reference)
#define TOK   2048            // B*S tokens
#define DIM   2048            // hidden size D
#define NEXP  8               // routed experts
#define HMID  1408            // moe_intermediate_size (shared expert = 2 pseudo-experts of HMID)
#define MT    16              // token tile (fallback kernel)
#define MTF   32              // token tile (fast kernel): each B-frag feeds 2 WMMAs
#define LGE   (HMID + 8)      // padded bf16 row stride of G tile in LDS

typedef __attribute__((ext_vector_type(16))) __bf16    v16bf;
typedef __attribute__((ext_vector_type(8)))  float     v8f;
typedef __attribute__((ext_vector_type(8)))  unsigned  v8u;

// ---------- fast fp32 helpers ----------
__device__ __forceinline__ float fast_rcp(float x) {
#if __has_builtin(__builtin_amdgcn_rcpf)
  return __builtin_amdgcn_rcpf(x);      // v_rcp_f32
#else
  return 1.f / x;
#endif
}

// ---------- fp32 -> bf16 (round to nearest even) ----------
__device__ __forceinline__ unsigned pk_bf16(float a, float b) {
#if __has_builtin(__builtin_amdgcn_cvt_pk_bf16_f32)
  typedef __attribute__((ext_vector_type(2))) __bf16 v2bf;
  v2bf r = __builtin_amdgcn_cvt_pk_bf16_f32(a, b);
  return __builtin_bit_cast(unsigned, r);
#else
  unsigned ua = __float_as_uint(a), ub = __float_as_uint(b);
  unsigned ra = (ua + 0x7FFFu + ((ua >> 16) & 1u)) >> 16;
  unsigned rb = (ub + 0x7FFFu + ((ub >> 16) & 1u)) >> 16;
  return (ra & 0xFFFFu) | (rb << 16);
#endif
}
__device__ __forceinline__ unsigned short bf16one(float a) {
  unsigned ua = __float_as_uint(a);
  return (unsigned short)((ua + 0x7FFFu + ((ua >> 16) & 1u)) >> 16);
}

// ---------- WMMA wrapper: D = A(16x32 bf16) * B(32x16 bf16) + C(f32) ----------
__device__ __forceinline__ v8f wmma_bf16(v16bf a, v16bf b, v8f c) {
  return __builtin_amdgcn_wmma_f32_16x16x32_bf16(
      false, a, false, b, (short)0, c, false, false);
}

// ---------- A fragment (16x32) from row-major f32 (fallback path) ----------
// lane = lo + 16*hi ; element e -> K = e + 16*hi ; M = lo
__device__ __forceinline__ v16bf load_A_g(const float* __restrict__ X, int ld,
                                          int row, int k0, int hi) {
  const float* p = X + (size_t)row * ld + (k0 + 16 * hi);
  v8u pk;
#pragma unroll
  for (int j = 0; j < 4; ++j) pk[j]     = pk_bf16(p[2 * j],      p[2 * j + 1]);
#pragma unroll
  for (int j = 0; j < 4; ++j) pk[4 + j] = pk_bf16(p[8 + 2 * j], p[8 + 2 * j + 1]);
  return __builtin_bit_cast(v16bf, pk);
}

// ---------- A fragment (16x32) from row-major bf16 in global (fast path) ----------
__device__ __forceinline__ v16bf load_A_bf(const unsigned short* __restrict__ ap, int k0) {
  const uint4* q0 = (const uint4*)(ap + k0);
  const uint4* q1 = (const uint4*)(ap + k0 + 8);
  uint4 a0 = *q0, a1 = *q1;
  v8u pk;
  pk[0] = a0.x; pk[1] = a0.y; pk[2] = a0.z; pk[3] = a0.w;
  pk[4] = a1.x; pk[5] = a1.y; pk[6] = a1.z; pk[7] = a1.w;
  return __builtin_bit_cast(v16bf, pk);
}

// ---------- A fragment (16x32) from LDS-resident bf16 G tile ----------
// row = absolute G row for this lane's M index (includes lo and group offset)
__device__ __forceinline__ v16bf load_A_lds(const unsigned* __restrict__ G,
                                            int row, int k0, int hi) {
  const uint4* q = (const uint4*)(G + (size_t)row * (LGE / 2) + ((k0 + 16 * hi) >> 1));
  uint4 a = q[0], b = q[1];
  v8u pk;
  pk[0] = a.x; pk[1] = a.y; pk[2] = a.z; pk[3] = a.w;
  pk[4] = b.x; pk[5] = b.y; pk[6] = b.z; pk[7] = b.w;
  return __builtin_bit_cast(v16bf, pk);
}

// ---------- per-wave LDS B staging (fallback path) ----------
__device__ __forceinline__ void stage_B(const float* __restrict__ W, int ld,
                                        int k0, int n0,
                                        unsigned short* __restrict__ st, int lane) {
  const float4* p = (const float4*)(W + (size_t)(k0 + lane) * ld + n0);
  float4 f0 = p[0], f1 = p[1], f2 = p[2], f3 = p[3];
  float v[16] = {f0.x, f0.y, f0.z, f0.w, f1.x, f1.y, f1.z, f1.w,
                 f2.x, f2.y, f2.z, f2.w, f3.x, f3.y, f3.z, f3.w};
#pragma unroll
  for (int n = 0; n < 16; ++n) st[n * 32 + lane] = bf16one(v[n]);
}
__device__ __forceinline__ v16bf read_B_stage(const unsigned short* __restrict__ st,
                                              int lo, int hi) {
  v8u pk = *(const v8u*)(st + lo * 32 + hi * 16);
  return __builtin_bit_cast(v16bf, pk);
}

// =====================================================================
// Kernel: gate -> softmax -> top-2 -> combine weights + expert token lists
// =====================================================================
__global__ void moe_gate_kernel(const float* __restrict__ X,
                                const float* __restrict__ GW,
                                float* __restrict__ comb,
                                int* __restrict__ cnt,
                                int* __restrict__ list) {
  int t = blockIdx.x * blockDim.x + threadIdx.x;
  if (t >= TOK) return;
  float acc[NEXP];
#pragma unroll
  for (int e = 0; e < NEXP; ++e) acc[e] = 0.f;
  const float* x = X + (size_t)t * DIM;
  for (int d = 0; d < DIM; ++d) {
    float xv = x[d];
#pragma unroll
    for (int e = 0; e < NEXP; ++e) acc[e] += xv * GW[d * NEXP + e];
  }
  float mx = acc[0];
#pragma unroll
  for (int e = 1; e < NEXP; ++e) mx = fmaxf(mx, acc[e]);
  float s = 0.f;
#pragma unroll
  for (int e = 0; e < NEXP; ++e) { acc[e] = __expf(acc[e] - mx); s += acc[e]; }
  float inv = fast_rcp(s);
#pragma unroll
  for (int e = 0; e < NEXP; ++e) acc[e] *= inv;
  int i1 = 0;
#pragma unroll
  for (int e = 1; e < NEXP; ++e) if (acc[e] > acc[i1]) i1 = e;
  int i2 = (i1 == 0) ? 1 : 0;
#pragma unroll
  for (int e = 0; e < NEXP; ++e) if (e != i1 && acc[e] > acc[i2]) i2 = e;
#pragma unroll
  for (int e = 0; e < NEXP; ++e)
    comb[(size_t)t * NEXP + e] = (e == i1 || e == i2) ? acc[e] : 0.f;
  int s1 = atomicAdd(&cnt[i1], 1); list[i1 * TOK + s1] = t;
  int s2 = atomicAdd(&cnt[i2], 1); list[i2 * TOK + s2] = t;
}

// =====================================================================
// Pre-pass kernels (fast path): fp32 -> bf16 once per call
// =====================================================================
__global__ void cvt_x_kernel(const float* __restrict__ X, unsigned short* __restrict__ Xb) {
  size_t i = ((size_t)blockIdx.x * blockDim.x + threadIdx.x) * 8;
  float4 a = *(const float4*)(X + i), b = *(const float4*)(X + i + 4);
  uint4 o;
  o.x = pk_bf16(a.x, a.y); o.y = pk_bf16(a.z, a.w);
  o.z = pk_bf16(b.x, b.y); o.w = pk_bf16(b.z, b.w);
  *(uint4*)(Xb + i) = o;
}

// Weights -> bf16 packed directly in WMMA B-fragment order.
// Tile (kt, nt) of a [Ktot x Ntot] row-major matrix -> 32 lanes x 32 bytes:
//   chunk[lane][e] = W[kt*32 + 16*hi + e][nt*16 + lo]
// Tiles laid out K-panel-major: tile index = nt*Ktiles + kt (per expert).
__global__ void pack_w_kernel(const float* __restrict__ W, unsigned short* __restrict__ P,
                              int ld, int Ktiles, int Ntiles,
                              unsigned long long w_estride) {
  long long gid = (long long)blockIdx.x * blockDim.x + threadIdx.x;
  long long tile = gid >> 5;
  int lane = (int)(gid & 31);
  long long tiles_per_e = (long long)Ktiles * Ntiles;
  long long e = tile / tiles_per_e;
  long long r = tile % tiles_per_e;
  int nt = (int)(r / Ktiles);
  int kt = (int)(r % Ktiles);
  int lo = lane & 15, hi = lane >> 4;
  const float* src = W + (unsigned long long)e * w_estride +
                     (size_t)(kt * 32 + 16 * hi) * ld + nt * 16 + lo;
  v8u pk;
#pragma unroll
  for (int j = 0; j < 8; ++j)
    pk[j] = pk_bf16(src[(size_t)(2 * j) * ld], src[(size_t)(2 * j + 1) * ld]);
  *(v8u*)(P + (tile * 32 + lane) * 16) = pk;
}

// =====================================================================
// FAST fused MLP: bf16 X + fragment-packed bf16 weights, MTF=32 tokens.
// Each B fragment (weights, the dominant stream) feeds 2 WMMAs.
// =====================================================================
__global__ __launch_bounds__(256) void moe_mlp_fast_kernel(
    const unsigned short* __restrict__ Xb,                    // [TOK, DIM] bf16
    const unsigned short* __restrict__ Bg,
    const unsigned short* __restrict__ Bu,
    unsigned long long gu_estride, int gu_ntstride,           // bf16-element strides
    const unsigned short* __restrict__ Bd,
    unsigned long long d_estride, int d_ntstride,
    const float* __restrict__ comb,
    const int* __restrict__ cnt, const int* __restrict__ list,
    float* __restrict__ Out) {
  __shared__ __align__(16) unsigned GsU[MTF * LGE / 2];       // ~90.6 KB bf16 G tile

  const int mtiles = TOK / MTF;
  const int expert = (int)blockIdx.x / mtiles;
  const int m0 = ((int)blockIdx.x % mtiles) * MTF;
  const int ntok = cnt ? cnt[expert] : TOK;
  if (m0 >= ntok) return;                                     // sparse dispatch

  const int wave = threadIdx.x >> 5;
  const int lane = threadIdx.x & 31;
  const int lo = lane & 15, hi = lane >> 4;

  // A-row tokens for the two 16-row groups
  int rowA0, rowA1;
  {
    int mA0 = m0 + lo, mA1 = m0 + 16 + lo;
    rowA0 = list ? ((mA0 < ntok) ? list[expert * TOK + mA0] : 0) : mA0;
    rowA1 = list ? ((mA1 < ntok) ? list[expert * TOK + mA1] : 0) : mA1;
  }
  const unsigned short* ap0 = Xb + (size_t)rowA0 * DIM + 16 * hi;
  const unsigned short* ap1 = Xb + (size_t)rowA1 * DIM + 16 * hi;

  int   rowO[2][8];
  float cw[2][8];
#pragma unroll
  for (int g = 0; g < 2; ++g)
#pragma unroll
    for (int r = 0; r < 8; ++r) {
      int m = m0 + g * 16 + r + 8 * hi;
      int valid = (m < ntok);
      int tk = list ? (valid ? list[expert * TOK + m] : 0) : m;
      rowO[g][r] = tk;
      cw[g][r] = comb ? (valid ? comb[(size_t)tk * NEXP + expert] : 0.f) : 1.f;
    }

  const unsigned short* bgE = Bg + (unsigned long long)expert * gu_estride + lane * 16;
  const unsigned short* buE = Bu + (unsigned long long)expert * gu_estride + lane * 16;
  const unsigned short* bdE = Bd + (unsigned long long)expert * d_estride + lane * 16;

  // ---- Phase 1: G = silu(X Wg) * (X Wu) * cw -> bf16 LDS ----
  for (int c = 0; c < HMID / 16; c += 8) {                    // 11 chunks of 128 cols
    int ntx = c + wave;
    const unsigned short* pg = bgE + (size_t)ntx * gu_ntstride;
    const unsigned short* pu = buE + (size_t)ntx * gu_ntstride;
    v8f cg0 = {}, cg1 = {}, cu0 = {}, cu1 = {};
    for (int kt = 0; kt < DIM / 32; ++kt) {
      v16bf a0 = load_A_bf(ap0, kt * 32);
      v16bf a1 = load_A_bf(ap1, kt * 32);
      v8u fg = *(const v8u*)pg; pg += 512;
      v8u fu = *(const v8u*)pu; pu += 512;
      v16bf bg = __builtin_bit_cast(v16bf, fg);
      v16bf bu = __builtin_bit_cast(v16bf, fu);
      cg0 = wmma_bf16(a0, bg, cg0);
      cg1 = wmma_bf16(a1, bg, cg1);
      cu0 = wmma_bf16(a0, bu, cu0);
      cu1 = wmma_bf16(a1, bu, cu1);
    }
    unsigned short* G16 = (unsigned short*)GsU;
    int n0 = ntx * 16;
#pragma unroll
    for (int r = 0; r < 8; ++r) {
      int m = r + 8 * hi;
      float g0 = cg0[r];
      float v0 = g0 * fast_rcp(1.f + __expf(-g0)) * cu0[r] * cw[0][r];
      G16[m * LGE + n0 + lo] = bf16one(v0);
      float g1 = cg1[r];
      float v1 = g1 * fast_rcp(1.f + __expf(-g1)) * cu1[r] * cw[1][r];
      G16[(m + 16) * LGE + n0 + lo] = bf16one(v1);
    }
  }
  __syncthreads();

  // ---- Phase 2: Out[tok,:] += G @ Wd ----
  for (int c = 0; c < DIM / 16; c += 8) {                     // 16 chunks of 128 cols
    int ntx = c + wave;
    const unsigned short* pd = bdE + (size_t)ntx * d_ntstride;
    v8f acc0 = {}, acc1 = {};
    for (int kt = 0; kt < HMID / 32; ++kt) {
      v16bf a0 = load_A_lds(GsU, lo, kt * 32, hi);
      v16bf a1 = load_A_lds(GsU, 16 + lo, kt * 32, hi);
      v8u fb = *(const v8u*)pd; pd += 512;
      v16bf b = __builtin_bit_cast(v16bf, fb);
      acc0 = wmma_bf16(a0, b, acc0);
      acc1 = wmma_bf16(a1, b, acc1);
    }
    int n0 = ntx * 16;
#pragma unroll
    for (int r = 0; r < 8; ++r) {
      unsafeAtomicAdd(Out + (size_t)rowO[0][r] * DIM + n0 + lo, acc0[r]);
      unsafeAtomicAdd(Out + (size_t)rowO[1][r] * DIM + n0 + lo, acc1[r]);
    }
  }
}

// =====================================================================
// FALLBACK fused MLP (small workspace): inline f32->bf16 staging via LDS.
// =====================================================================
__global__ __launch_bounds__(256) void moe_mlp_kernel(
    const float* __restrict__ X,
    const float* __restrict__ Wg, const float* __restrict__ Wu,
    unsigned long long wgu_estride, int ldw,
    const float* __restrict__ Wd, unsigned long long wd_estride,
    const float* __restrict__ comb,
    const int* __restrict__ cnt, const int* __restrict__ list,
    float* __restrict__ Out) {
  __shared__ __align__(16) unsigned GsU[MT * LGE / 2];
  __shared__ __align__(32) unsigned short Bst[8][2][16 * 32];

  const int mtiles = TOK / MT;
  const int expert = (int)blockIdx.x / mtiles;
  const int m0 = ((int)blockIdx.x % mtiles) * MT;
  const int ntok = cnt ? cnt[expert] : TOK;
  if (m0 >= ntok) return;

  const int wave = threadIdx.x >> 5;
  const int lane = threadIdx.x & 31;
  const int lo = lane & 15, hi = lane >> 4;

  const float* wg = Wg + (unsigned long long)expert * wgu_estride;
  const float* wu = Wu + (unsigned long long)expert * wgu_estride;
  const float* wd = Wd + (unsigned long long)expert * wd_estride;
  unsigned short* stg = &Bst[wave][0][0];
  unsigned short* stu = &Bst[wave][1][0];

  const int mA = m0 + lo;
  const int rowA = list ? ((mA < ntok) ? list[expert * TOK + mA] : 0) : mA;

  int   rowO[8];
  float cw[8];
#pragma unroll
  for (int r = 0; r < 8; ++r) {
    int m = m0 + r + 8 * hi;
    int valid = (m < ntok);
    int tk = list ? (valid ? list[expert * TOK + m] : 0) : m;
    rowO[r] = tk;
    cw[r] = comb ? (valid ? comb[(size_t)tk * NEXP + expert] : 0.f) : 1.f;
  }

  for (int h0 = 0; h0 < HMID; h0 += 128) {
    int n0 = h0 + wave * 16;
    v8f cg = {}, cu = {};
    for (int k0 = 0; k0 < DIM; k0 += 32) {
      stage_B(wg, ldw, k0, n0, stg, lane);
      stage_B(wu, ldw, k0, n0, stu, lane);
      v16bf a  = load_A_g(X, DIM, rowA, k0, hi);
      v16bf bg = read_B_stage(stg, lo, hi);
      v16bf bu = read_B_stage(stu, lo, hi);
      cg = wmma_bf16(a, bg, cg);
      cu = wmma_bf16(a, bu, cu);
    }
    unsigned short* G16 = (unsigned short*)GsU;
#pragma unroll
    for (int r = 0; r < 8; ++r) {
      int m = r + 8 * hi;
      float g = cg[r];
      float val = g * fast_rcp(1.f + __expf(-g)) * cu[r] * cw[r];
      G16[m * LGE + n0 + lo] = bf16one(val);
    }
  }
  __syncthreads();

  for (int d0 = 0; d0 < DIM; d0 += 128) {
    int n0 = d0 + wave * 16;
    v8f c = {};
    for (int k0 = 0; k0 < HMID; k0 += 32) {
      stage_B(wd, DIM, k0, n0, stg, lane);
      v16bf a = load_A_lds(GsU, lo, k0, hi);
      v16bf b = read_B_stage(stg, lo, hi);
      c = wmma_bf16(a, b, c);
    }
#pragma unroll
    for (int r = 0; r < 8; ++r)
      unsafeAtomicAdd(Out + (size_t)rowO[r] * DIM + n0 + lo, c[r]);
  }
}

// =====================================================================
extern "C" void kernel_launch(void* const* d_in, const int* in_sizes, int n_in,
                              void* d_out, int out_size, void* d_ws, size_t ws_size,
                              hipStream_t stream) {
  (void)in_sizes; (void)n_in; (void)out_size;
  const float* X   = (const float*)d_in[0];
  const float* GW  = (const float*)d_in[1];
  const float* Wg  = (const float*)d_in[2];
  const float* Wu  = (const float*)d_in[3];
  const float* Wd  = (const float*)d_in[4];
  const float* SWg = (const float*)d_in[5];
  const float* SWu = (const float*)d_in[6];
  const float* SWd = (const float*)d_in[7];
  float* Out = (float*)d_out;
  char* ws = (char*)d_ws;

  // workspace layout (bytes)
  const size_t OFF_COMB = 0;                         // TOK*NEXP f32   (64 KB)
  const size_t OFF_CNT  = 262144;                    // NEXP ints
  const size_t OFF_LIST = 262400;                    // NEXP*TOK ints  (64 KB)
  const size_t OFF_XB   = 1ull << 20;                // bf16 X         (8 MB)
  const size_t SZ_GU_R  = 46137344;                  // 8*64*88*512 bf16
  const size_t SZ_SH    = 11534336;                  // 64*176*512 bf16
  const size_t OFF_BGR  = OFF_XB + 8388608;
  const size_t OFF_BUR  = OFF_BGR + SZ_GU_R;
  const size_t OFF_BDR  = OFF_BUR + SZ_GU_R;
  const size_t OFF_BSG  = OFF_BDR + SZ_GU_R;
  const size_t OFF_BSU  = OFF_BSG + SZ_SH;
  const size_t OFF_BSD  = OFF_BSU + SZ_SH;
  const size_t NEED     = OFF_BSD + SZ_SH;           // ~174 MB

  float* comb = (float*)(ws + OFF_COMB);
  int*   cnt  = (int*)(ws + OFF_CNT);
  int*   list = (int*)(ws + OFF_LIST);

  hipMemsetAsync(Out, 0, (size_t)TOK * DIM * sizeof(float), stream);
  hipMemsetAsync(cnt, 0, 256, stream);
  moe_gate_kernel<<<TOK / 256, 256, 0, stream>>>(X, GW, comb, cnt, list);

  if (ws_size >= NEED) {
    // ---------------- fast path: bf16 pre-pack ----------------
    unsigned short* Xb  = (unsigned short*)(ws + OFF_XB);
    unsigned short* BgR = (unsigned short*)(ws + OFF_BGR);
    unsigned short* BuR = (unsigned short*)(ws + OFF_BUR);
    unsigned short* BdR = (unsigned short*)(ws + OFF_BDR);
    unsigned short* Bsg = (unsigned short*)(ws + OFF_BSG);
    unsigned short* Bsu = (unsigned short*)(ws + OFF_BSU);
    unsigned short* Bsd = (unsigned short*)(ws + OFF_BSD);

    cvt_x_kernel<<<(TOK * DIM / 8) / 256, 256, 0, stream>>>(X, Xb);
    pack_w_kernel<<<8 * 64 * 88 * 32 / 256, 256, 0, stream>>>(
        Wg, BgR, HMID, 64, 88, (unsigned long long)DIM * HMID);
    pack_w_kernel<<<8 * 64 * 88 * 32 / 256, 256, 0, stream>>>(
        Wu, BuR, HMID, 64, 88, (unsigned long long)DIM * HMID);
    pack_w_kernel<<<8 * 44 * 128 * 32 / 256, 256, 0, stream>>>(
        Wd, BdR, DIM, 44, 128, (unsigned long long)HMID * DIM);
    pack_w_kernel<<<64 * 176 * 32 / 256, 256, 0, stream>>>(
        SWg, Bsg, 2 * HMID, 64, 176, 0ull);
    pack_w_kernel<<<64 * 176 * 32 / 256, 256, 0, stream>>>(
        SWu, Bsu, 2 * HMID, 64, 176, 0ull);
    pack_w_kernel<<<88 * 128 * 32 / 256, 256, 0, stream>>>(
        SWd, Bsd, DIM, 88, 128, 0ull);

    // shared expert = 2 pseudo-experts (weight 1, identity tokens)
    moe_mlp_fast_kernel<<<(TOK / MTF) * 2, 256, 0, stream>>>(
        Xb, Bsg, Bsu, 88ull * 64 * 512, 64 * 512,
        Bsd, 44ull * 512, 88 * 512,
        nullptr, nullptr, nullptr, Out);
    // routed experts, sparse-dispatched
    moe_mlp_fast_kernel<<<(TOK / MTF) * NEXP, 256, 0, stream>>>(
        Xb, BgR, BuR, 64ull * 88 * 512, 64 * 512,
        BdR, 44ull * 128 * 512, 44 * 512,
        comb, cnt, list, Out);
  } else {
    // ---------------- fallback: inline staging ----------------
    moe_mlp_kernel<<<(TOK / MT) * 2, 256, 0, stream>>>(
        X, SWg, SWu, (unsigned long long)HMID, 2 * HMID,
        SWd, (unsigned long long)HMID * DIM,
        nullptr, nullptr, nullptr, Out);
    moe_mlp_kernel<<<(TOK / MT) * NEXP, 256, 0, stream>>>(
        X, Wg, Wu, (unsigned long long)DIM * HMID, HMID,
        Wd, (unsigned long long)HMID * DIM,
        comb, cnt, list, Out);
  }
}
